// MoeLayer_37984690765955
// MI455X (gfx1250) — matrix operand order repaired and checked by reading, not compiled, not run on hardware
//
#include <hip/hip_runtime.h>
#include <cmath>

// ---------------- problem constants (match reference) ----------------
#define TOK     8192      // B*N = 4*2048
#define DMODEL  1024
#define NEXP    8
#define TILE_M  64        // tokens per FFN block
#define NCH     32        // output-column chunk staged per weight slab
#define NCHUNKS (DMODEL / NCH)            // 32
#define TILES_PER_E (TOK / TILE_M)        // 128 (worst case: all tokens -> one expert)
#define SLAB_U4 (NCH * DMODEL / 8)        // 4096 u32x4 per slab
#define SLAB_PER_THREAD (SLAB_U4 / 256)   // 16 u32x4 = 64 VGPRs

// ---------------- types (trivial ext-vectors: union-safe) ----------------
typedef __attribute__((ext_vector_type(16))) __bf16      v16bf;
typedef __attribute__((ext_vector_type(8)))  float       v8f;
typedef __attribute__((ext_vector_type(4)))  unsigned int u32x4;
typedef __attribute__((ext_vector_type(2)))  unsigned int u32x2;
typedef __attribute__((ext_vector_type(4)))  float       f32x4;

union BF16Frag {
    struct { u32x4 lo, hi; } u;
    v16bf v;
};

__device__ __forceinline__ unsigned short f2bf_bits(float f) {
    union { float f; unsigned u; } c; c.f = f;
    unsigned r = (c.u + 0x7FFFu + ((c.u >> 16) & 1u)) >> 16;  // RNE
    return (unsigned short)r;
}
__device__ __forceinline__ __bf16 f2bf(float f) {
    union { unsigned short s; __bf16 b; } c; c.s = f2bf_bits(f);
    return c.b;
}
// Branch-free GELU: x*sigmoid(1.5957691*x + 0.0713548*x^3) (tanh-GELU identity).
// Error vs exact-erf GELU is below the bf16 storage quantization of h.
__device__ __forceinline__ float gelu_fast(float v) {
    float u = v * (1.5957691216057308f + 0.07135481627f * v * v);
    return v / (1.0f + __expf(-u));
}
// 16-element bf16 fragment (two ds_load_b128) per the CDNA5 16-bit operand
// layout: elems 0..7 = K {base..base+7}, elems 8..15 = K {base+16..base+23}.
__device__ __forceinline__ v16bf ld_frag(const __bf16* p) {
    BF16Frag r;
    r.u.lo = *(const u32x4*)p;
    r.u.hi = *(const u32x4*)(p + 16);
    return r.v;
}
// register-double-buffered weight slab staging (64 VGPRs/thread)
__device__ __forceinline__ void slab_load(const __bf16* base, int tid, u32x4 r[SLAB_PER_THREAD]) {
    const u32x4* src = (const u32x4*)base;
#pragma unroll
    for (int i = 0; i < SLAB_PER_THREAD; ++i) r[i] = src[i * 256 + tid];
}
__device__ __forceinline__ void slab_store(__bf16* Ws, int tid, const u32x4 r[SLAB_PER_THREAD]) {
    u32x4* dst = (u32x4*)Ws;
#pragma unroll
    for (int i = 0; i < SLAB_PER_THREAD; ++i) dst[i * 256 + tid] = r[i];
}

// ---------------- kernel 1: f32 [E][K][N] -> bf16 [E][N][K] ----------------
__global__ __launch_bounds__(256)
void moe_transpose_w(const float* __restrict__ w, __bf16* __restrict__ wT) {
    __shared__ float tile[32][33];
    const int e  = blockIdx.z;
    const float* src = w  + (size_t)e * DMODEL * DMODEL;
    __bf16*      dst = wT + (size_t)e * DMODEL * DMODEL;
    const int k0 = blockIdx.x * 32;
    const int n0 = blockIdx.y * 32;
    const int tx = threadIdx.x;          // 0..31
    const int ty = threadIdx.y;          // 0..7
    for (int i = 0; i < 32; i += 8)      // coalesced along n
        tile[ty + i][tx] = src[(size_t)(k0 + ty + i) * DMODEL + n0 + tx];
    __syncthreads();
    for (int i = 0; i < 32; i += 8)      // coalesced along k
        dst[(size_t)(n0 + ty + i) * DMODEL + k0 + tx] = f2bf(tile[tx][ty + i]);
}

// ---------------- kernel 2: router (one wave32 per token) ----------------
__global__ __launch_bounds__(256)
void moe_router(const float* __restrict__ x, const float* __restrict__ gate_w,
                __bf16* __restrict__ xb, int* __restrict__ counts,
                int* __restrict__ tok_list, float* __restrict__ wgt_list) {
    const int lane = threadIdx.x & 31;
    const int t    = blockIdx.x * 8 + (threadIdx.x >> 5);   // token id
    const float* xr = x + (size_t)t * DMODEL;

    float acc[NEXP];
#pragma unroll
    for (int e = 0; e < NEXP; ++e) acc[e] = 0.0f;

#pragma unroll
    for (int c = 0; c < 8; ++c) {
        const int d = c * 128 + lane * 4;
        f32x4 xv = *(const f32x4*)(xr + d);
        u32x2 pk;                                   // bf16 copy of x, read-once
        pk.x = (unsigned)f2bf_bits(xv.x) | ((unsigned)f2bf_bits(xv.y) << 16);
        pk.y = (unsigned)f2bf_bits(xv.z) | ((unsigned)f2bf_bits(xv.w) << 16);
        *(u32x2*)(xb + (size_t)t * DMODEL + d) = pk;
#pragma unroll
        for (int e = 0; e < NEXP; ++e) {
            f32x4 g = *(const f32x4*)(gate_w + (size_t)e * DMODEL + d);
            acc[e] += xv.x * g.x + xv.y * g.y + xv.z * g.z + xv.w * g.w;
        }
    }
#pragma unroll
    for (int e = 0; e < NEXP; ++e)
#pragma unroll
        for (int off = 16; off > 0; off >>= 1)
            acc[e] += __shfl_xor(acc[e], off, 32);

    if (lane == 0) {
        float mx = acc[0];
#pragma unroll
        for (int e = 1; e < NEXP; ++e) mx = fmaxf(mx, acc[e]);
        float p[NEXP], s = 0.0f;
#pragma unroll
        for (int e = 0; e < NEXP; ++e) { p[e] = __expf(acc[e] - mx); s += p[e]; }
        const float inv = 1.0f / s;
        int e0 = 0; float p0 = p[0];
#pragma unroll
        for (int e = 1; e < NEXP; ++e) if (p[e] > p0) { p0 = p[e]; e0 = e; }
        int e1 = -1; float p1 = -1.0f;
#pragma unroll
        for (int e = 0; e < NEXP; ++e) if (e != e0 && p[e] > p1) { p1 = p[e]; e1 = e; }
        int s0 = atomicAdd(&counts[e0], 1);
        tok_list[e0 * TOK + s0] = t;  wgt_list[e0 * TOK + s0] = p0 * inv;
        int s1 = atomicAdd(&counts[e1], 1);
        tok_list[e1 * TOK + s1] = t;  wgt_list[e1 * TOK + s1] = p1 * inv;
    }
}

// ---------------- kernel 3: grouped expert FFN with WMMA ----------------
// LDS: Xs 64x1024 bf16 (128KB) | Hs 64x1024 bf16 (128KB) | Ws 32x1024 bf16 (64KB) = 320KB
// Weight slabs are register-double-buffered; all epilogue global loads (bias,
// token ids, combine weights) are issued BEFORE the WMMA k-loop so their
// latency hides under matrix work.
__global__ __launch_bounds__(256)
void moe_ffn(const __bf16* __restrict__ xb,
             const __bf16* __restrict__ w1T, const float* __restrict__ b1,
             const __bf16* __restrict__ w2T, const float* __restrict__ b2,
             const int* __restrict__ counts,
             const int* __restrict__ tok_list, const float* __restrict__ wgt_list,
             float* __restrict__ out) {
    const int e    = blockIdx.x / TILES_PER_E;
    const int m0   = (blockIdx.x % TILES_PER_E) * TILE_M;
    const int cnt  = counts[e];
    if (m0 >= cnt) return;                       // uniform early-exit
    int rows = cnt - m0; if (rows > TILE_M) rows = TILE_M;

    extern __shared__ char smem[];
    __bf16* Xs = (__bf16*)smem;                  // [TILE_M][DMODEL]
    __bf16* Hs = Xs + TILE_M * DMODEL;           // [TILE_M][DMODEL]
    __bf16* Ws = Hs + TILE_M * DMODEL;           // [NCH][DMODEL] (transposed weights)

    const int tid  = threadIdx.x;
    const int lane = tid & 31;
    const int wid  = tid >> 5;                   // 0..7
    const int mi   = wid & 3;                    // 4 M-bands of 16
    const int nj   = wid >> 2;                   // 2 N-subtiles of 16 per chunk

    const __bf16* w1e = w1T + (size_t)e * DMODEL * DMODEL;
    const __bf16* w2e = w2T + (size_t)e * DMODEL * DMODEL;

    u32x4 wreg[SLAB_PER_THREAD];
    slab_load(w1e, tid, wreg);                   // slab 0 in flight during X gather

    // ---- gather token rows (bf16) into LDS; zero-fill padding rows ----
    {
        const int r = tid >> 2;                  // 0..63
        const int c = tid & 3;                   // 4 threads per row
        u32x4* dst = (u32x4*)(Xs + r * DMODEL) + c * 32;
        if (r < rows) {
            const int tok = tok_list[e * TOK + m0 + r];
            const u32x4* src = (const u32x4*)(xb + (size_t)tok * DMODEL) + c * 32;
#pragma unroll 4
            for (int i = 0; i < 32; ++i) dst[i] = src[i];
        } else {
            u32x4 z = (u32x4)0;
#pragma unroll 4
            for (int i = 0; i < 32; ++i) dst[i] = z;
        }
    }
    __syncthreads();

    // CDNA5 16-bit operand lane addressing
    const int am   = mi * 16 + (lane & 15);      // A row in tile
    const int koff = (lane >> 4) * 8;            // lanes 0-15: K{0..7,16..23}; 16-31: K{8..15,24..31}
    const int nl   = nj * 16 + (lane & 15);      // B column -> row of transposed slab
    const int csub = (lane >> 4) * 8;            // C: lanes 16-31 hold M rows +8

    // ================= GEMM1: h = gelu(X @ W1 + b1) =================
    for (int nc = 0; nc < NCHUNKS; ++nc) {
        const int n0 = nc * NCH;
        slab_store(Ws, tid, wreg);               // commit prefetched slab
        __syncthreads();
        if (nc + 1 < NCHUNKS)                    // prefetch next slab into regs
            slab_load(w1e + (size_t)(n0 + NCH) * DMODEL, tid, wreg);
        else
            slab_load(w2e, tid, wreg);           // prefetch GEMM2 slab 0 across the seam

        const int ng   = n0 + nj * 16 + (lane & 15);
        const float bv = b1[e * DMODEL + ng];    // issued before k-loop: latency hidden

        v8f acc = {};
        const __bf16* Ap = Xs + am * DMODEL + koff;
        const __bf16* Bp = Ws + nl * DMODEL + koff;
#pragma unroll 8
        for (int k = 0; k < DMODEL; k += 32) {
            v16bf A  = ld_frag(Ap + k);
            v16bf Bv = ld_frag(Bp + k);
            acc = __builtin_amdgcn_wmma_f32_16x16x32_bf16(
                      false, A, false, Bv, (short)0, acc, false, false);
        }
#pragma unroll
        for (int r = 0; r < 8; ++r)
            Hs[(mi * 16 + csub + r) * DMODEL + ng] = f2bf(gelu_fast(acc[r] + bv));
        __syncthreads();                         // Ws consumed; Hs chunk complete
    }

    // ================= GEMM2: out += combine * (H @ W2 + b2) =================
    for (int gc = 0; gc < NCHUNKS; ++gc) {
        const int g0 = gc * NCH;
        slab_store(Ws, tid, wreg);
        __syncthreads();
        if (gc + 1 < NCHUNKS)
            slab_load(w2e + (size_t)(g0 + NCH) * DMODEL, tid, wreg);

        const int gg   = g0 + nj * 16 + (lane & 15);
        const float bv = b2[e * DMODEL + gg];    // hoisted epilogue loads
        int   tk[8];
        float wv[8];
#pragma unroll
        for (int r = 0; r < 8; ++r) {
            const int m = mi * 16 + csub + r;
            tk[r] = (m < rows) ? tok_list[e * TOK + m0 + m] : -1;
            wv[r] = (m < rows) ? wgt_list[e * TOK + m0 + m] : 0.0f;
        }

        v8f acc = {};
        const __bf16* Ap = Hs + am * DMODEL + koff;
        const __bf16* Bp = Ws + nl * DMODEL + koff;
#pragma unroll 8
        for (int k = 0; k < DMODEL; k += 32) {
            v16bf A  = ld_frag(Ap + k);
            v16bf Bv = ld_frag(Bp + k);
            acc = __builtin_amdgcn_wmma_f32_16x16x32_bf16(
                      false, A, false, Bv, (short)0, acc, false, false);
        }
#pragma unroll
        for (int r = 0; r < 8; ++r) {
            if (tk[r] >= 0)
                atomicAdd(out + (size_t)tk[r] * DMODEL + gg, wv[r] * (acc[r] + bv));
        }
        __syncthreads();
    }
}

// ---------------- host launcher ----------------
extern "C" void kernel_launch(void* const* d_in, const int* in_sizes, int n_in,
                              void* d_out, int out_size, void* d_ws, size_t ws_size,
                              hipStream_t stream) {
    const float* x      = (const float*)d_in[0];   // [B,N,D]
    const float* gate_w = (const float*)d_in[1];   // [E,D]
    const float* w1     = (const float*)d_in[2];   // [E,D,D]  (in,out)
    const float* b1     = (const float*)d_in[3];   // [E,D]
    const float* w2     = (const float*)d_in[4];   // [E,D,D]
    const float* b2     = (const float*)d_in[5];   // [E,D]
    float* out = (float*)d_out;

    // workspace layout
    constexpr size_t XB_OFF   = 0;
    constexpr size_t XB_SZ    = (size_t)TOK * DMODEL * 2;             // 16 MB bf16 x
    constexpr size_t W1T_OFF  = XB_OFF + XB_SZ;
    constexpr size_t WT_SZ    = (size_t)NEXP * DMODEL * DMODEL * 2;   // 16 MB each
    constexpr size_t W2T_OFF  = W1T_OFF + WT_SZ;
    constexpr size_t CNT_OFF  = W2T_OFF + WT_SZ;
    constexpr size_t CNT_SZ   = 256;
    constexpr size_t TOKL_OFF = CNT_OFF + CNT_SZ;
    constexpr size_t TOKL_SZ  = (size_t)NEXP * TOK * 4;
    constexpr size_t WGTL_OFF = TOKL_OFF + TOKL_SZ;

    char* ws = (char*)d_ws;
    __bf16* xb       = (__bf16*)(ws + XB_OFF);
    __bf16* w1T      = (__bf16*)(ws + W1T_OFF);
    __bf16* w2T      = (__bf16*)(ws + W2T_OFF);
    int*    counts   = (int*)   (ws + CNT_OFF);
    int*    tok_list = (int*)   (ws + TOKL_OFF);
    float*  wgt_list = (float*) (ws + WGTL_OFF);

    hipMemsetAsync(counts, 0, CNT_SZ, stream);
    hipMemsetAsync(out, 0, (size_t)out_size * sizeof(float), stream);

    // 1) transpose + bf16-convert expert weights: [E][K][N] -> [E][N][K]
    dim3 tb(32, 8, 1), tg(DMODEL / 32, DMODEL / 32, NEXP);
    moe_transpose_w<<<tg, tb, 0, stream>>>(w1, w1T);
    moe_transpose_w<<<tg, tb, 0, stream>>>(w2, w2T);

    // 2) router (+ bf16 conversion of x): one wave per token
    moe_router<<<TOK / 8, 256, 0, stream>>>(x, gate_w, xb, counts, tok_list, wgt_list);

    // 3) grouped expert FFN, 320KB dynamic LDS per block
    constexpr size_t LDS_BYTES = (size_t)(2 * TILE_M + NCH) * DMODEL * 2;  // 327680
    moe_ffn<<<NEXP * TILES_PER_E, 256, LDS_BYTES, stream>>>(
        xb, w1T, b1, w2T, b2, counts, tok_list, wgt_list, out);
}
